// SelfAttention_651
// MI455X (gfx1250) — matrix operand order repaired and checked by reading, not compiled, not run on hardware
//
#include <hip/hip_runtime.h>
#include <hip/hip_bf16.h>

// CDNA5 / gfx1250, wave32 only.
typedef _Float16 h8   __attribute__((ext_vector_type(8)));
typedef _Float16 v16h __attribute__((ext_vector_type(16)));
typedef float    v8f  __attribute__((ext_vector_type(8)));
typedef unsigned int u32x4 __attribute__((ext_vector_type(4)));
typedef int          i32x8 __attribute__((ext_vector_type(8)));
typedef int          i32x4 __attribute__((ext_vector_type(4)));

#define Lq 256   // seq length
#define Ff 64    // n_fields (K of projections)
#define Dd 256   // field-info dim

// LDS layout (bytes).  U32 staging (64KB) overlays the Q region, which is
// only written after the f32->f16 conversion pass has consumed the staging.
#define OFF_Q    0          // 256*256*2 = 131072   (bytes [0,131072))
#define OFF_U32  0          // 256*64*4  = 65536    (dead before Q written)
#define OFF_K    131072     // 256*256*2 = 131072
#define OFF_U16  262144     // 256*64*2  = 32768
#define OFF_VBAR 294912     // 256*4     = 1024
#define OFF_WV   295936     // 64*4 + 4
#define SMEM_BYTES 296448

// ---- fragment loaders (contiguous k order, wave32) -------------------

// A-matrix 16x32 f16 from row-major [row][k] LDS, 'stride' in elements.
// lane<16 : M=lane,    K = 0..15
// lane>=16: M=lane-16, K = 16..31
__device__ __forceinline__ v16h load_a_frag(const _Float16* lds, int row,
                                            int stride, int kbase, int lane) {
  int kh = (lane >> 4) & 1;
  const _Float16* p = lds + row * stride + kbase + kh * 16;
  h8 lo = *(const h8*)p;        // K = kbase + kh*16 + [0..7]
  h8 hi = *(const h8*)(p + 8);  // K = kbase + kh*16 + 8 + [0..7]
  return __builtin_shufflevector(lo, hi, 0,1,2,3,4,5,6,7,8,9,10,11,12,13,14,15);
}

// B-matrix 32x16 f16 from [n][k] row-major LDS (i.e. B transposed storage).
// lane<16 : N=lane, K=0..15 ; lane>=16 : N=lane-16, K=16..31  (contiguous)
__device__ __forceinline__ v16h load_b_frag(const _Float16* lds, int nrow,
                                            int stride, int kbase, int lane) {
  const _Float16* p = lds + nrow * stride + kbase + ((lane >> 4) & 1) * 16;
  return *(const v16h*)p;       // 32 bytes contiguous
}

// B-fragment built from a global f32 weight matrix stored [d][f] row-major:
// this is exactly [n][k] storage for B[k=f][n=d].
__device__ __forceinline__ v16h load_wfrag_global(const float* W, int d,
                                                  int kbase, int lane) {
  const float* p = W + d * Ff + kbase + ((lane >> 4) & 1) * 16;
  v16h b;
#pragma unroll
  for (int i = 0; i < 16; ++i) b[i] = (_Float16)p[i];
  return b;
}

__device__ __forceinline__ v8f wmma16(const v16h& a, const v16h& b, const v8f& c) {
  return __builtin_amdgcn_wmma_f32_16x16x32_f16(false, a, false, b,
                                                (short)0, c, false, false);
}

// ---- TDM: DMA one batch of U (64KB f32, contiguous) into LDS -------------
// D# per ISA 8.3/8.4: 2D tensor, one row of 16384 4-byte elements.
__device__ __forceinline__ void tdm_load_u(const float* gsrc, void* ldsdst) {
  unsigned long long ga = (unsigned long long)(uintptr_t)gsrc;
  unsigned int lds_off  = (unsigned int)(uintptr_t)ldsdst;  // addr[31:0] = LDS offset
  u32x4 g0;
  g0[0] = 1u;                                        // count=1 (valid), user D#
  g0[1] = lds_off;                                   // lds_addr
  g0[2] = (unsigned int)(ga & 0xFFFFFFFFu);          // global_addr[31:0]
  g0[3] = (unsigned int)((ga >> 32) & 0x01FFFFFFu)   // global_addr[56:32]
        | (2u << 30);                                // type = 2 ("image")
  i32x8 g1;
  g1[0] = (2 << 16);                // data_size = 2 -> 4 bytes; wg_mask = 0
  g1[1] = (int)(16384u << 16);      // tensor_dim0[15:0] -> bits[63:48]
  g1[2] = (1 << 16);                // tensor_dim0[31:16]=0; tensor_dim1 = 1
  g1[3] = (int)(16384u << 16);      // tile_dim0 = 16384 -> bits[127:112]
  g1[4] = 1;                        // tile_dim1 = 1; tile_dim2 = 0
  g1[5] = 16384;                    // tensor_dim0_stride[31:0]
  g1[6] = 0;                        // stride msbs / dim1_stride
  g1[7] = 0;
  i32x4 z4 = {0, 0, 0, 0};          // groups 2/3 unused (<=2D tensor)
  i32x8 z8 = {0, 0, 0, 0, 0, 0, 0, 0};
  __builtin_amdgcn_tensor_load_to_lds(g0, g1, z4, z4, z8, 0);
}

// ---- prep: wvbar[f] = mean_d Wv[d,f], bvbar = mean(bv) into d_ws ----------
__global__ void prep_vbar_kernel(const float* __restrict__ Wv,
                                 const float* __restrict__ bv,
                                 float* __restrict__ ws) {
  int f = threadIdx.x;  // 64 threads
  float s = 0.f;
  for (int d = 0; d < Dd; ++d) s += Wv[d * Ff + f];
  ws[f] = s * (1.0f / (float)Dd);
  if (f == 0) {
    float sb = 0.f;
    for (int d = 0; d < Dd; ++d) sb += bv[d];
    ws[Ff] = sb * (1.0f / (float)Dd);
  }
}

// ---- main: one workgroup per batch ---------------------------------------
__global__ void __launch_bounds__(256)
attn_field_kernel(const float* __restrict__ U,
                  const float* __restrict__ Wq, const float* __restrict__ bq,
                  const float* __restrict__ Wk, const float* __restrict__ bk,
                  const float* __restrict__ wsv,   // [wvbar(64), bvbar]
                  const int*   __restrict__ fidxp,
                  float* __restrict__ out) {
  extern __shared__ char smem[];
  _Float16* Qlds = (_Float16*)(smem + OFF_Q);    // [256][256]
  float*    U32  = (float*)   (smem + OFF_U32);  // staging, overlays Qlds
  _Float16* Klds = (_Float16*)(smem + OFF_K);    // [256][256] (m-major = B layout)
  _Float16* Ulds = (_Float16*)(smem + OFF_U16);  // [256][64]
  float*    vbar = (float*)   (smem + OFF_VBAR); // [256]
  float*    wvl  = (float*)   (smem + OFF_WV);   // [64] + bvbar at [64]

  const int b    = blockIdx.x;
  const int tid  = threadIdx.x;
  const int wave = tid >> 5;        // 0..7
  const int lane = tid & 31;
  const int ln   = lane & 15;
  const int kh   = lane >> 4;
  const int fidx = fidxp[0];

  // --- kick off the TDM copy of U_b (wave-uniform branch: wave 0 only) ----
  if (wave == 0) {
    tdm_load_u(U + (size_t)b * Lq * Ff, (void*)U32);
  }

  // --- per-wave weight fragments, held in registers (overlaps DMA) --------
  // wave owns d-tiles {2*wave, 2*wave+1}
  v16h wqf[2][2], wkf[2][2];
  float bqv[2], bkv[2];
#pragma unroll
  for (int dti = 0; dti < 2; ++dti) {
    int d = (2 * wave + dti) * 16 + ln;
#pragma unroll
    for (int s = 0; s < 2; ++s) {
      wqf[dti][s] = load_wfrag_global(Wq, d, 32 * s, lane);
      wkf[dti][s] = load_wfrag_global(Wk, d, 32 * s, lane);
    }
    bqv[dti] = bq[d];
    bkv[dti] = bk[d];
  }
  if (tid < Ff + 1) wvl[tid] = wsv[tid];  // wvbar + bvbar

  if (wave == 0) __builtin_amdgcn_s_wait_tensorcnt(0);
  __syncthreads();   // U32 staging + wvl visible to all

  // --- convert staged U_b to f16 row-major [l][f] -------------------------
  {
    const float4* src = (const float4*)(U32 + tid * Ff);
    _Float16* dst = Ulds + tid * Ff;
#pragma unroll
    for (int i = 0; i < 16; ++i) {
      float4 v = src[i];
      dst[4 * i + 0] = (_Float16)v.x;
      dst[4 * i + 1] = (_Float16)v.y;
      dst[4 * i + 2] = (_Float16)v.z;
      dst[4 * i + 3] = (_Float16)v.w;
    }
  }
  __syncthreads();   // U16 ready; U32 staging now dead (Q may overwrite)

  // --- vbar[m] = U[m,:] . wvbar + bvbar  (one row per thread) -------------
  {
    float s = wvl[Ff];
    const _Float16* ur = Ulds + tid * Ff;
#pragma unroll 8
    for (int f = 0; f < Ff; ++f) s += (float)ur[f] * wvl[f];
    vbar[tid] = s;
  }

  // --- phase 1: Q = U Wq^T + bq ; K = U Wk^T + bk  (WMMA, f32 accum) ------
  for (int lt = 0; lt < 16; ++lt) {
    int arow = lt * 16 + ln;
    v8f accQ[2], accK[2];
#pragma unroll
    for (int dti = 0; dti < 2; ++dti)
#pragma unroll
      for (int r = 0; r < 8; ++r) { accQ[dti][r] = bqv[dti]; accK[dti][r] = bkv[dti]; }

#pragma unroll
    for (int s = 0; s < 2; ++s) {
      v16h a = load_a_frag(Ulds, arow, Ff, 32 * s, lane);
      accQ[0] = wmma16(a, wqf[0][s], accQ[0]);
      accQ[1] = wmma16(a, wqf[1][s], accQ[1]);
      accK[0] = wmma16(a, wkf[0][s], accK[0]);
      accK[1] = wmma16(a, wkf[1][s], accK[1]);
    }
    // C layout: lane<16 -> rows lt*16+r (col d), lane>=16 -> rows lt*16+8+r
    int rbase = lt * 16 + kh * 8;
#pragma unroll
    for (int dti = 0; dti < 2; ++dti) {
      int d = (2 * wave + dti) * 16 + ln;
#pragma unroll
      for (int r = 0; r < 8; ++r) {
        Qlds[(rbase + r) * Dd + d] = (_Float16)accQ[dti][r];
        Klds[(rbase + r) * Dd + d] = (_Float16)accK[dti][r];
      }
    }
  }
  __syncthreads();

  // --- phase 2: S = Q K^T / sqrt(F); softmax-weighted sum of vbar ---------
  const float scale = 0.125f;  // 1/sqrt(64)
#pragma unroll
  for (int lti = 0; lti < 2; ++lti) {
    int lt = 2 * wave + lti;
    int arow = lt * 16 + ln;

    // Q A-fragments for this row-tile are invariant across the mt loop:
    // hoist all 8 k-steps into registers (64 VGPRs; weights are dead now).
    v16h afr[8];
#pragma unroll
    for (int s = 0; s < 8; ++s) afr[s] = load_a_frag(Qlds, arow, Dd, 32 * s, lane);

    float accN[8], accD[8];
#pragma unroll
    for (int r = 0; r < 8; ++r) { accN[r] = 0.f; accD[r] = 0.f; }

    for (int mt = 0; mt < 16; ++mt) {
      int brow = mt * 16 + ln;
      v8f c = {};
#pragma unroll
      for (int s = 0; s < 8; ++s) {
        v16h bb = load_b_frag(Klds, brow, Dd, 32 * s, lane);
        c = wmma16(afr[s], bb, c);
      }
      float vb = vbar[mt * 16 + ln];  // lane's column m
#pragma unroll
      for (int r = 0; r < 8; ++r) {
        float e = __expf(c[r] * scale);
        accD[r] += e;
        accN[r] += e * vb;
      }
    }
    // reduce over the 16 lanes holding the tile's 16 columns (stays in half)
#pragma unroll
    for (int r = 0; r < 8; ++r) {
#pragma unroll
      for (int off = 1; off < 16; off <<= 1) {
        accN[r] += __shfl_xor(accN[r], off, 32);
        accD[r] += __shfl_xor(accD[r], off, 32);
      }
    }
    if (ln == 0) {  // lanes 0 (rows 0-7) and 16 (rows 8-15) write
      int rbase = lt * 16 + kh * 8;
#pragma unroll
      for (int r = 0; r < 8; ++r) {
        int row = rbase + r;
        float o = accN[r] / accD[r];
        const float* urow = U + ((size_t)b * Lq + row) * Ff;
        float add;
        if (fidx > 0) {
          add = urow[fidx];
        } else {
          float su = 0.f;
          for (int f = 0; f < Ff; ++f) su += urow[f];
          add = su * (1.0f / (float)Ff);
        }
        out[(size_t)b * Lq + row] = o + add;
      }
    }
  }
}

extern "C" void kernel_launch(void* const* d_in, const int* in_sizes, int n_in,
                              void* d_out, int out_size, void* d_ws, size_t ws_size,
                              hipStream_t stream) {
  const float* U  = (const float*)d_in[0];
  const float* Wq = (const float*)d_in[1];
  const float* bq = (const float*)d_in[2];
  const float* Wk = (const float*)d_in[3];
  const float* bk = (const float*)d_in[4];
  const float* Wv = (const float*)d_in[5];
  const float* bv = (const float*)d_in[6];
  const int* fidx = (const int*)d_in[7];
  float* out = (float*)d_out;
  float* ws  = (float*)d_ws;

  const int B = in_sizes[0] / (Lq * Ff);

  (void)hipFuncSetAttribute((const void*)attn_field_kernel,
                            hipFuncAttributeMaxDynamicSharedMemorySize,
                            SMEM_BYTES);

  prep_vbar_kernel<<<1, 64, 0, stream>>>(Wv, bv, ws);
  attn_field_kernel<<<B, 256, SMEM_BYTES, stream>>>(U, Wq, bq, Wk, bk, ws, fidx, out);
  (void)n_in; (void)out_size; (void)ws_size;
}
